// HGTNet_51960514347439
// MI455X (gfx1250) — matrix operand order repaired and checked by reading, not compiled, not run on hardware
//
#include <hip/hip_runtime.h>
#include <math.h>

#define NNODES 30000
#define NEDGES 480000
#define FIN    16
#define HID    64
#define HEADS  2
#define DHEAD  32
#define NEG_SLOPE 0.2f

typedef __attribute__((ext_vector_type(16))) _Float16 v16h;
typedef __attribute__((ext_vector_type(8)))  float    v8f;

// ---------------------------------------------------------------------------
// Weight pre-pack: rearrange W[K,NF] (f32 row-major) into per-lane WMMA
// B-fragments (f16), so the GEMM loads one contiguous 32B v16h per fragment.
// Fragment (s,t), lane l, element j  <-  W[s*32 + 8*(l>>4) + (j&7) + 16*(j>>3)][t*16 + (l&15)]
// ---------------------------------------------------------------------------
__global__ void pack_w(const float* __restrict__ W, _Float16* __restrict__ Wp,
                       int K, int NF)
{
  int tid = blockIdx.x * blockDim.x + threadIdx.x;
  int NT = NF >> 4;
  int total = ((K + 31) / 32) * NT * 32 * 16;
  if (tid >= total) return;
  int j    = tid & 15;
  int lane = (tid >> 4) & 31;
  int f    = tid >> 9;
  int s = f / NT, t = f % NT;
  int kk   = s * 32 + 8 * (lane >> 4) + (j & 7) + ((j >> 3) << 4);
  int ncol = t * 16 + (lane & 15);
  Wp[tid] = (kk < K) ? (_Float16)W[(size_t)kk * NF + ncol] : (_Float16)0.0f;
}

// ---------------------------------------------------------------------------
// Fused GEMM: out[M,NF] = act(A[M,K] * W[K,NF] + bias).
// Compile-time K/NF/ACT -> fully unrolled, no spills.
// ACT: 0=none 1=relu 2=tanh. COLSUM: accumulate activated column sums only.
// One wave owns a 16-row x NF-col tile. M must be a multiple of 16.
// ---------------------------------------------------------------------------
template<int K, int NF, int ACT, bool COLSUM>
__global__ void gemm_wmma(const float* __restrict__ A,
                          const _Float16* __restrict__ Wp,
                          const float* __restrict__ bias,
                          float* __restrict__ out,
                          float* __restrict__ colsum,
                          int M)
{
  constexpr int KSTEPS = (K + 31) / 32;
  constexpr int NT     = NF >> 4;
  const int lane    = threadIdx.x & 31;
  const int wave    = threadIdx.x >> 5;
  const int rowTile = blockIdx.x * (blockDim.x >> 5) + wave;
  if (rowTile * 16 >= M) return;
  const int m    = lane & 15;
  const int half = lane >> 4;
  const int row  = rowTile * 16 + m;

  v8f acc[NT];
#pragma unroll
  for (int t = 0; t < NT; ++t)
#pragma unroll
    for (int r = 0; r < 8; ++r) acc[t][r] = 0.0f;

  const float4* A4 = (const float4*)(A + (size_t)row * K);

#pragma unroll
  for (int s = 0; s < KSTEPS; ++s) {
    const int k0 = s * 32;
    v16h a;
    // halfs 0..7 : K = k0 + 8*half + j   (always valid: K >= k0+16 here)
    {
      float4 c0 = A4[(k0 + 8 * half) >> 2];
      float4 c1 = A4[((k0 + 8 * half) >> 2) + 1];
      a[0] = (_Float16)c0.x; a[1] = (_Float16)c0.y;
      a[2] = (_Float16)c0.z; a[3] = (_Float16)c0.w;
      a[4] = (_Float16)c1.x; a[5] = (_Float16)c1.y;
      a[6] = (_Float16)c1.z; a[7] = (_Float16)c1.w;
    }
    // halfs 8..15 : K = k0 + 16 + 8*half + j (valid iff K >= k0+32; folds)
    if (k0 + 32 <= K) {
      float4 c2 = A4[(k0 + 16 + 8 * half) >> 2];
      float4 c3 = A4[((k0 + 16 + 8 * half) >> 2) + 1];
      a[8]  = (_Float16)c2.x; a[9]  = (_Float16)c2.y;
      a[10] = (_Float16)c2.z; a[11] = (_Float16)c2.w;
      a[12] = (_Float16)c3.x; a[13] = (_Float16)c3.y;
      a[14] = (_Float16)c3.z; a[15] = (_Float16)c3.w;
    } else {
#pragma unroll
      for (int j = 8; j < 16; ++j) a[j] = (_Float16)0.0f;
    }
#pragma unroll
    for (int t = 0; t < NT; ++t) {
      v16h b = *(const v16h*)(Wp + (((size_t)s * NT + t) * 32 + lane) * 16);
      acc[t] = __builtin_amdgcn_wmma_f32_16x16x32_f16(
          false, a, false, b, (short)0, acc[t], false, false);
    }
  }

#pragma unroll
  for (int t = 0; t < NT; ++t) {
    const int ncol = t * 16 + m;
    const float bv = bias[ncol];
    float cs = 0.0f;
#pragma unroll
    for (int r = 0; r < 8; ++r) {
      float v = acc[t][r] + bv;                    // D row = r + 8*half
      if (ACT == 1)      v = v > 0.0f ? v : 0.0f;
      else if (ACT == 2) v = tanhf(v);
      if (!COLSUM)
        out[(size_t)(rowTile * 16 + r + 8 * half) * NF + ncol] = v;
      else
        cs += v;
    }
    if (COLSUM) atomicAdd(&colsum[ncol], cs);
  }
}

// ---------------------------------------------------------------------------
// Per-node attention logits: a[n,h] = dot(p[n,h,:], att[h,:])
// ---------------------------------------------------------------------------
__global__ void node_att(const float* __restrict__ p, const float* __restrict__ att,
                         float* __restrict__ outa, int n)
{
  int tid = blockIdx.x * blockDim.x + threadIdx.x;
  if (tid >= n * HEADS) return;
  int node = tid >> 1, h = tid & 1;
  const float* pr = p + (size_t)node * HID + h * DHEAD;
  const float* av = att + h * DHEAD;
  float s = 0.0f;
#pragma unroll
  for (int d = 0; d < DHEAD; ++d) s += pr[d] * av[d];
  outa[tid] = s;
}

__global__ void fillk(float* __restrict__ p, float v, int n)
{
  int i = blockIdx.x * blockDim.x + threadIdx.x;
  if (i < n) p[i] = v;
}

// order-preserving float<->uint encoding so segment-max can use atomicMax(u32)
__device__ __forceinline__ unsigned fenc(float f) {
  unsigned u = __float_as_uint(f);
  return (u & 0x80000000u) ? ~u : (u | 0x80000000u);
}
__device__ __forceinline__ float fdec(unsigned u) {
  return __uint_as_float((u & 0x80000000u) ? (u ^ 0x80000000u) : ~u);
}

__global__ void edge_alpha_max(const int* __restrict__ ei,
                               const float* __restrict__ as,
                               const float* __restrict__ ad,
                               unsigned* __restrict__ amax)
{
  int tid = blockIdx.x * blockDim.x + threadIdx.x;
  if (tid >= NEDGES * HEADS) return;
  int e = tid >> 1, h = tid & 1;
  int s = ei[e], d = ei[NEDGES + e];
  float a = as[s * HEADS + h] + ad[d * HEADS + h];
  a = a > 0.0f ? a : NEG_SLOPE * a;               // leaky relu
  atomicMax(&amax[d * HEADS + h], fenc(a));
}

__global__ void edge_exp_sum(const int* __restrict__ ei,
                             const float* __restrict__ as,
                             const float* __restrict__ ad,
                             const unsigned* __restrict__ amax,
                             float* __restrict__ den,
                             float* __restrict__ exw)
{
  int tid = blockIdx.x * blockDim.x + threadIdx.x;
  if (tid >= NEDGES * HEADS) return;
  int e = tid >> 1, h = tid & 1;
  int s = ei[e], d = ei[NEDGES + e];
  float a = as[s * HEADS + h] + ad[d * HEADS + h];
  a = a > 0.0f ? a : NEG_SLOPE * a;
  float ex = expf(a - fdec(amax[d * HEADS + h]));
  exw[tid] = ex;
  atomicAdd(&den[d * HEADS + h], ex);
}

__global__ void edge_scatter(const int* __restrict__ ei,
                             const float* __restrict__ psrc,
                             const float* __restrict__ exw,
                             const float* __restrict__ den,
                             float* __restrict__ outb)
{
  long long tid = (long long)blockIdx.x * blockDim.x + threadIdx.x;
  if (tid >= (long long)NEDGES * HID) return;
  int e = (int)(tid >> 6), f = (int)(tid & 63);
  int h = f >> 5;
  int s = ei[e], d = ei[NEDGES + e];
  float w = exw[e * HEADS + h] / (den[d * HEADS + h] + 1e-16f);
  atomicAdd(&outb[(size_t)d * HID + f], psrc[(size_t)s * HID + f] * w);
}

__global__ void relu_inplace(float* __restrict__ p, int n)
{
  int i = blockIdx.x * blockDim.x + threadIdx.x;
  if (i < n) { float v = p[i]; p[i] = v > 0.0f ? v : 0.0f; }
}

__global__ void sem_attn(const float* __restrict__ ks, const float* __restrict__ q,
                         float* __restrict__ attn, int T, float invN)
{
  if (threadIdx.x != 0 || blockIdx.x != 0) return;
  float s[2]; float mx = -1e30f;
  for (int t = 0; t < T; ++t) {
    float a = 0.0f;
    for (int f = 0; f < HID; ++f) a += q[f] * ks[t * HID + f] * invN;
    s[t] = a; mx = fmaxf(mx, a);
  }
  float dsum = 0.0f;
  for (int t = 0; t < T; ++t) { s[t] = expf(s[t] - mx); dsum += s[t]; }
  for (int t = 0; t < T; ++t) attn[t] = s[t] / dsum;
}

__global__ void combine2(const float* __restrict__ a, const float* __restrict__ b,
                         const float* __restrict__ attn, float* __restrict__ o, int n)
{
  int i = blockIdx.x * blockDim.x + threadIdx.x;
  if (i < n) o[i] = attn[0] * a[i] + attn[1] * b[i];
}

__global__ void copyk(const float* __restrict__ a, float* __restrict__ o, int n)
{
  int i = blockIdx.x * blockDim.x + threadIdx.x;
  if (i < n) o[i] = a[i];
}

__global__ void init_value(float* __restrict__ dst, const float* __restrict__ b)
{
  if (threadIdx.x == 0 && blockIdx.x == 0) dst[0] = b[0];
}

__global__ void value_reduce(const float* __restrict__ x, const float* __restrict__ w,
                             float* __restrict__ acc, int rows, float scale)
{
  int r = blockIdx.x * blockDim.x + threadIdx.x;
  if (r >= rows) return;
  float s = 0.0f;
#pragma unroll
  for (int j = 0; j < HID; ++j) s += x[(size_t)r * HID + j] * w[j];
  atomicAdd(acc, s * scale);
}

// ---------------------------------------------------------------------------
extern "C" void kernel_launch(void* const* d_in, const int* in_sizes, int n_in,
                              void* d_out, int out_size, void* d_ws, size_t ws_size,
                              hipStream_t stream)
{
  const float* x_job    = (const float*)d_in[0];
  const float* x_mach   = (const float*)d_in[1];
  const int*   ei_jm    = (const int*)d_in[2];
  const int*   ei_mj    = (const int*)d_in[3];
  const int*   ei_jj    = (const int*)d_in[4];
  const float* enc_w_j  = (const float*)d_in[5];
  const float* enc_b_j  = (const float*)d_in[6];
  const float* enc_w_m  = (const float*)d_in[7];
  const float* enc_b_m  = (const float*)d_in[8];
  const float* proj_w   = (const float*)d_in[9];
  const float* proj_b   = (const float*)d_in[10];
  const float* att_src  = (const float*)d_in[11];
  const float* att_dst  = (const float*)d_in[12];
  const float* klin_w   = (const float*)d_in[13];
  const float* klin_b   = (const float*)d_in[14];
  const float* q_sem    = (const float*)d_in[15];
  const float* lin0_w   = (const float*)d_in[16];
  const float* lin0_b   = (const float*)d_in[17];
  const float* linout_w = (const float*)d_in[18];
  const float* linout_b = (const float*)d_in[19];
  const float* linv_w   = (const float*)d_in[20];
  const float* linv_b   = (const float*)d_in[21];
  float* out = (float*)d_out;

  // ---- workspace carve-up (floats) ----
  float* ws = (float*)d_ws;
  const size_t NH = (size_t)NNODES * HID;      // 1,920,000
  float* xj     = ws;            float* xm     = xj + NH;
  float* pj     = xm + NH;       float* pm     = pj + NH;
  float* out_jm = pm + NH;       float* out_mj = out_jm + NH;
  float* out_jj = out_mj + NH;
  float* hbuf   = out_jj + NH;                         // 2N*HID
  float* asb    = hbuf + 2 * NH;                       // 6 x N*HEADS
  float* as_jm = asb, *ad_jm = asb + 1*(size_t)NNODES*HEADS;
  float* as_mj = asb + 2*(size_t)NNODES*HEADS, *ad_mj = asb + 3*(size_t)NNODES*HEADS;
  float* as_jj = asb + 4*(size_t)NNODES*HEADS, *ad_jj = asb + 5*(size_t)NNODES*HEADS;
  float* amax = asb + 6*(size_t)NNODES*HEADS;          // N*HEADS (uint-encoded)
  float* den  = amax + (size_t)NNODES*HEADS;           // N*HEADS
  float* exw  = den  + (size_t)NNODES*HEADS;           // E*HEADS
  float* ksum = exw  + (size_t)NEDGES*HEADS;           // 2*HID
  float* attn = ksum + 2*HID;                          // pad 8 -> keeps 32B align
  // ---- packed f16 weight fragments (32B-aligned: float offset % 8 == 0) ----
  _Float16* wp = (_Float16*)(attn + 8);
  const size_t FR1664 = 1 * 4 * 32 * 16;   // 2048 halfs  (K=16,NF=64)
  const size_t FR6464 = 2 * 4 * 32 * 16;   // 4096 halfs  (K=64,NF=64)
  const size_t FR6416 = 2 * 1 * 32 * 16;   // 1024 halfs  (K=64,NF=16)
  _Float16* wp_enc_j = wp;                       // 2048
  _Float16* wp_enc_m = wp_enc_j + FR1664;        // 2048
  _Float16* wp_proj  = wp_enc_m + FR1664;        // 4 x 4096 (l0j,l0m,l1j,l1m)
  _Float16* wp_klin  = wp_proj + 4 * FR6464;     // 2 x 4096
  _Float16* wp_lin0  = wp_klin + 2 * FR6464;     // 4096
  _Float16* wp_lout  = wp_lin0 + FR6464;         // 1024

  const int TPB = 256;
  auto gb = [](long long n, int t){ return (unsigned)((n + t - 1) / t); };
  const unsigned GEMM_BLKS = gb((NNODES + 15) / 16, 8);     // 16 rows/wave, 8 waves/blk
  const unsigned GEMM_BLKS2 = gb((2 * NNODES + 15) / 16, 8);

  // ---- pack all weight matrices into WMMA B-fragment layout (f16) ----
  pack_w<<<gb(FR1664, TPB), TPB, 0, stream>>>(enc_w_j, wp_enc_j, FIN, HID);
  pack_w<<<gb(FR1664, TPB), TPB, 0, stream>>>(enc_w_m, wp_enc_m, FIN, HID);
  for (int i = 0; i < 4; ++i)
    pack_w<<<gb(FR6464, TPB), TPB, 0, stream>>>(proj_w + (size_t)i*HID*HID,
                                                wp_proj + (size_t)i*FR6464, HID, HID);
  for (int l = 0; l < 2; ++l)
    pack_w<<<gb(FR6464, TPB), TPB, 0, stream>>>(klin_w + (size_t)l*HID*HID,
                                                wp_klin + (size_t)l*FR6464, HID, HID);
  pack_w<<<gb(FR6464, TPB), TPB, 0, stream>>>(lin0_w, wp_lin0, HID, HID);
  pack_w<<<gb(FR6416, TPB), TPB, 0, stream>>>(linout_w, wp_lout, HID, 16);

  auto run_edge = [&](const float* psrc, const int* ei,
                      const float* as, const float* ad, float* outb) {
    fillk<<<gb(NNODES*HEADS, TPB), TPB, 0, stream>>>(amax, 0.0f, NNODES*HEADS); // enc(-NaN) ~ -inf
    fillk<<<gb(NNODES*HEADS, TPB), TPB, 0, stream>>>(den, 0.0f, NNODES*HEADS);
    edge_alpha_max<<<gb((long long)NEDGES*HEADS, TPB), TPB, 0, stream>>>(
        ei, as, ad, (unsigned*)amax);
    edge_exp_sum<<<gb((long long)NEDGES*HEADS, TPB), TPB, 0, stream>>>(
        ei, as, ad, (const unsigned*)amax, den, exw);
    fillk<<<gb(NH, TPB), TPB, 0, stream>>>(outb, 0.0f, (int)NH);
    edge_scatter<<<gb((long long)NEDGES*HID, TPB), TPB, 0, stream>>>(
        ei, psrc, exw, den, outb);
    relu_inplace<<<gb(NH, TPB), TPB, 0, stream>>>(outb, (int)NH);
  };

  // ---- encoder: relu(x @ W + b) ----
  gemm_wmma<FIN, HID, 1, false><<<GEMM_BLKS, 256, 0, stream>>>(
      x_job, wp_enc_j, enc_b_j, xj, nullptr, NNODES);
  gemm_wmma<FIN, HID, 1, false><<<GEMM_BLKS, 256, 0, stream>>>(
      x_mach, wp_enc_m, enc_b_m, xm, nullptr, NNODES);

  for (int l = 0; l < 2; ++l) {
    gemm_wmma<HID, HID, 0, false><<<GEMM_BLKS, 256, 0, stream>>>(
        xj, wp_proj + (size_t)(l*2+0)*FR6464, proj_b + (size_t)(l*2+0)*HID,
        pj, nullptr, NNODES);
    gemm_wmma<HID, HID, 0, false><<<GEMM_BLKS, 256, 0, stream>>>(
        xm, wp_proj + (size_t)(l*2+1)*FR6464, proj_b + (size_t)(l*2+1)*HID,
        pm, nullptr, NNODES);

    const float* at_s = att_src + (size_t)l*3*HEADS*DHEAD;
    const float* at_d = att_dst + (size_t)l*3*HEADS*DHEAD;
    unsigned na = gb((long long)NNODES*HEADS, TPB);
    node_att<<<na, TPB, 0, stream>>>(pj, at_s + 0*HEADS*DHEAD, as_jm, NNODES);
    node_att<<<na, TPB, 0, stream>>>(pm, at_d + 0*HEADS*DHEAD, ad_jm, NNODES);
    node_att<<<na, TPB, 0, stream>>>(pm, at_s + 1*HEADS*DHEAD, as_mj, NNODES);
    node_att<<<na, TPB, 0, stream>>>(pj, at_d + 1*HEADS*DHEAD, ad_mj, NNODES);
    node_att<<<na, TPB, 0, stream>>>(pj, at_s + 2*HEADS*DHEAD, as_jj, NNODES);
    node_att<<<na, TPB, 0, stream>>>(pj, at_d + 2*HEADS*DHEAD, ad_jj, NNODES);

    run_edge(pj, ei_jm, as_jm, ad_jm, out_jm);   // j -> m
    run_edge(pm, ei_mj, as_mj, ad_mj, out_mj);   // m -> j
    run_edge(pj, ei_jj, as_jj, ad_jj, out_jj);   // j -> j

    // semantic attention for xj over {out_mj, out_jj}; colsum-fused tanh GEMM
    const float* klb = klin_b + (size_t)l*HID;
    const float* q   = q_sem  + (size_t)l*HID;
    _Float16* klw    = wp_klin + (size_t)l*FR6464;
    fillk<<<1, 128, 0, stream>>>(ksum, 0.0f, 2*HID);
    gemm_wmma<HID, HID, 2, true><<<GEMM_BLKS, 256, 0, stream>>>(
        out_mj, klw, klb, nullptr, ksum, NNODES);
    gemm_wmma<HID, HID, 2, true><<<GEMM_BLKS, 256, 0, stream>>>(
        out_jj, klw, klb, nullptr, ksum + HID, NNODES);
    sem_attn<<<1, 1, 0, stream>>>(ksum, q, attn, 2, 1.0f / NNODES);
    combine2<<<gb(NH, TPB), TPB, 0, stream>>>(out_mj, out_jj, attn, xj, (int)NH);
    // xm group has T=1 -> softmax over one element == identity
    copyk<<<gb(NH, TPB), TPB, 0, stream>>>(out_jm, xm, (int)NH);
  }

  // ---- heads ----
  const long long actN = 2LL * NNODES * 16;
  init_value<<<1, 1, 0, stream>>>(out + actN, linv_b);
  value_reduce<<<gb(NNODES, TPB), TPB, 0, stream>>>(xj, linv_w, out + actN, NNODES,
                                                    1.0f / (2.0f * NNODES));
  value_reduce<<<gb(NNODES, TPB), TPB, 0, stream>>>(xm, linv_w, out + actN, NNODES,
                                                    1.0f / (2.0f * NNODES));
  gemm_wmma<HID, HID, 0, false><<<GEMM_BLKS, 256, 0, stream>>>(
      xj, wp_lin0, lin0_b, hbuf, nullptr, NNODES);
  gemm_wmma<HID, HID, 0, false><<<GEMM_BLKS, 256, 0, stream>>>(
      xm, wp_lin0, lin0_b, hbuf + NH, nullptr, NNODES);
  gemm_wmma<HID, 16, 2, false><<<GEMM_BLKS2, 256, 0, stream>>>(
      hbuf, wp_lout, linout_b, out, nullptr, 2 * NNODES);
}
